// Decoder_25769803776501
// MI455X (gfx1250) — compile-verified, matrix operand order checked by reading, not run on hardware
//
#include <hip/hip_runtime.h>
#include <hip/hip_bf16.h>

// Problem sizes (compile-time, from the reference).
#define B_DIM 512
#define L_DIM 64     // n_latent
#define H_DIM 32     // hidden
#define G_DIM 5000   // genes of interest
#define K_DIM 64     // per-gene K
#define BN_EPS 1e-5f

typedef __attribute__((ext_vector_type(16))) _Float16 v16h;
typedef __attribute__((ext_vector_type(8)))  float    v8f;
typedef __attribute__((ext_vector_type(4)))  float    v4f;

// ---------------------------------------------------------------------------
// Kernel 1: h = BN(ReLU(latent @ W1 + b1)).  [512,32] -- tiny.
// Writes f32 copy (for rho) and f16 copy (WMMA A operand source) to workspace.
// ---------------------------------------------------------------------------
__global__ __launch_bounds__(256) void mlp_kernel(
    const float* __restrict__ latent, const float* __restrict__ W1,
    const float* __restrict__ b1, const float* __restrict__ gamma,
    const float* __restrict__ beta, const float* __restrict__ mean,
    const float* __restrict__ var, float* __restrict__ h32,
    _Float16* __restrict__ h16)
{
    int tid = blockIdx.x * 256 + threadIdx.x;
    if (tid >= B_DIM * H_DIM) return;
    int b = tid >> 5;        // batch row
    int j = tid & 31;        // hidden unit
    float s = b1[j];
    #pragma unroll 8
    for (int l = 0; l < L_DIM; ++l)
        s = fmaf(latent[b * L_DIM + l], W1[l * H_DIM + j], s);
    s = fmaxf(s, 0.0f);                                        // ReLU
    s = gamma[j] * (s - mean[j]) * rsqrtf(var[j] + BN_EPS) + beta[j];
    h32[tid] = s;
    h16[tid] = (_Float16)s;
}

// ---------------------------------------------------------------------------
// Kernel 2: logit[b, gi, k] = sum_h h[b,h] * lw[genes_oi[gi], h, k]
// One wave per gene. B-fragments (32x64 weights -> 4 x v16h) loaded once and
// held in registers across all 32 batch tiles. One v_wmma_f32_16x16x32_f16
// per 16x16 output tile (WMMA K == H == 32, fully consumed in one op).
//
// Output-column permutation for wide stores: B-fragment n, lane column ln
// holds lw column (4*ln + n).  Then acc[0..3][v] in a lane are 4 CONSECUTIVE
// output columns -> one global_store_b128 per (v): 8 b128 stores per batch
// tile, each covering two dense 256B segments.
//
// ISA register layouts used (CDNA5 ISA):
//   16-bit A 16x32:  lane<16 -> M=lane,   halves: K=0..7 then K=16..23
//                    lane>=16-> M=lane-16,halves: K=8..15 then K=24..31
//   16-bit B 32x16:  lane indexes column N (lane&15), K striped over halves
//                    exactly as for A (symmetric operand layout).
//   32-bit D 16x16:  VGPR v -> row (v + 8*(lane>=16)), lane&15 -> column.
// ---------------------------------------------------------------------------
__global__ __launch_bounds__(256) void logit_wmma_kernel(
    const float* __restrict__ lw,          // [N_GENES, 32, 64] f32
    const int* __restrict__ genes_oi,      // [G_DIM]
    const _Float16* __restrict__ h16,      // [512, 32] f16
    float* __restrict__ out_logit)         // [512, G_DIM, 64] f32
{
    const int wave = threadIdx.x >> 5;
    const int lane = threadIdx.x & 31;
    const int gi   = blockIdx.x * 8 + wave;     // gene index (0..4999), exact
    if (gi >= G_DIM) return;                    // wave-uniform; EXEC stays all-1s

    const int g = genes_oi[gi];
    const float* lwg = lw + (size_t)g * (H_DIM * K_DIM);

    const int hi    = lane >> 4;      // 0: lanes 0-15, 1: lanes 16-31
    const int kbase = hi * 8;         // K offset within each group of 16
    const int ln    = lane & 15;

    // ---- B fragments: lw column (4*ln + n) for all 32 h, f32 -> f16 ----
    v16h bfrag[4];
    #pragma unroll
    for (int n = 0; n < 4; ++n) {
        const int col = 4 * ln + n;
        #pragma unroll
        for (int i = 0; i < 8; ++i) {
            bfrag[n][i]     = (_Float16)lwg[(kbase + i) * K_DIM + col];       // K=kbase..+7
            bfrag[n][i + 8] = (_Float16)lwg[(16 + kbase + i) * K_DIM + col];  // K=16+kbase..
        }
    }

    // ---- sweep the 32 batch tiles, reusing B fragments from registers ----
    for (int bt = 0; bt < 32; ++bt) {
        const int row = bt * 16 + ln;
        // Two 16-byte loads fill the A fragment in the ISA layout.
        v16h afrag;
        ((uint4*)&afrag)[0] = *(const uint4*)(h16 + row * H_DIM + kbase);
        ((uint4*)&afrag)[1] = *(const uint4*)(h16 + row * H_DIM + 16 + kbase);

        v8f acc[4];
        #pragma unroll
        for (int n = 0; n < 4; ++n) {
            v8f c = {};
            acc[n] = __builtin_amdgcn_wmma_f32_16x16x32_f16(
                false, afrag, false, bfrag[n], (short)0, c, false, false);
        }

        // ---- store D tiles: one b128 per accumulator row, non-temporal ----
        const int m0 = bt * 16 + hi * 8;
        #pragma unroll
        for (int v = 0; v < 8; ++v) {
            v4f pack;
            pack.x = acc[0][v];
            pack.y = acc[1][v];
            pack.z = acc[2][v];
            pack.w = acc[3][v];
            const size_t off =
                ((size_t)(m0 + v) * G_DIM + gi) * (size_t)K_DIM + 4 * ln;
            __builtin_nontemporal_store(pack, (v4f*)(out_logit + off));
        }
    }
}

// ---------------------------------------------------------------------------
// Kernel 3: rho[b, gi] = dot(h[b,:], rw[genes_oi[gi],:]).  10 MB output,
// all reads L2-resident -- a flat VALU kernel is ample.
// ---------------------------------------------------------------------------
__global__ __launch_bounds__(256) void rho_kernel(
    const float* __restrict__ h32, const int* __restrict__ genes_oi,
    const float* __restrict__ rw, float* __restrict__ out_rho)
{
    size_t tid = (size_t)blockIdx.x * 256 + threadIdx.x;
    if (tid >= (size_t)B_DIM * G_DIM) return;
    const int b  = (int)(tid / G_DIM);
    const int gi = (int)(tid % G_DIM);
    const int g  = genes_oi[gi];
    const float* hrow = h32 + b * H_DIM;
    const float* w    = rw + (size_t)g * H_DIM;
    float s = 0.0f;
    #pragma unroll
    for (int h = 0; h < H_DIM; ++h) s = fmaf(hrow[h], w[h], s);
    __builtin_nontemporal_store(s, out_rho + tid);
}

// ---------------------------------------------------------------------------
extern "C" void kernel_launch(void* const* d_in, const int* in_sizes, int n_in,
                              void* d_out, int out_size, void* d_ws, size_t ws_size,
                              hipStream_t stream)
{
    const float* latent   = (const float*)d_in[0];
    const int*   genes_oi = (const int*)d_in[1];   // harness: integer -> int*
    const float* W1       = (const float*)d_in[2];
    const float* b1       = (const float*)d_in[3];
    const float* bn_gamma = (const float*)d_in[4];
    const float* bn_beta  = (const float*)d_in[5];
    const float* bn_mean  = (const float*)d_in[6];
    const float* bn_var   = (const float*)d_in[7];
    const float* lw       = (const float*)d_in[8];
    const float* rw       = (const float*)d_in[9];

    float* out_logit = (float*)d_out;                                  // [512,5000,64]
    float* out_rho   = out_logit + (size_t)B_DIM * G_DIM * K_DIM;      // [512,5000]

    // Workspace: h16 [512*32] f16 (32 KB) then h32 [512*32] f32 (64 KB).
    _Float16* h16 = (_Float16*)d_ws;
    float*    h32 = (float*)((char*)d_ws + (size_t)B_DIM * H_DIM * sizeof(_Float16));

    // 1) shared MLP head (tiny)
    mlp_kernel<<<(B_DIM * H_DIM + 255) / 256, 256, 0, stream>>>(
        latent, W1, b1, bn_gamma, bn_beta, bn_mean, bn_var, h32, h16);

    // 2) logit GEMM: 5000 waves (one gene each), 8 waves/block -> 625 blocks
    logit_wmma_kernel<<<G_DIM / 8, 256, 0, stream>>>(lw, genes_oi, h16, out_logit);

    // 3) rho: 512*5000 = 2,560,000 threads -> 10000 blocks exactly
    rho_kernel<<<(B_DIM * G_DIM) / 256, 256, 0, stream>>>(h32, genes_oi, rw, out_rho);
}
